// AttentionLayer_39728447488156
// MI455X (gfx1250) — matrix operand order repaired and checked by reading, not compiled, not run on hardware
//
#include <hip/hip_runtime.h>
#include <math.h>

// ---------------------------------------------------------------------------
// AttentionLayer on MI455X (gfx1250, wave32, WMMA).
//   q = E@Wq^T+bq ; k = E@Wk^T+bk ; v = E@Wv^T+bv ; out = softmax(qk^T)v
// N=8192, D=1024, fp32 in/out. ~326 GFLOP vs ~100 MB traffic -> compute
// bound: all GEMMs on v_wmma_f32_16x16x32_f16 (f32 accum). E/W pre-converted
// to f16; Q/K row-major f16; V stored transposed (contiguous P@V B-frags).
// Flash-style fusion (scores never hit memory). Hot k-loops fully unrolled:
// each row's k-walk spans only 2 KB, so every load is base + immediate
// offset (no per-iteration address VALU -> no WMMA->VALU hazard nops), with
// ping-pong register buffer sets so loads clause up and overlap WMMAs.
// ---------------------------------------------------------------------------

typedef __attribute__((ext_vector_type(16))) _Float16 v16h;
typedef __attribute__((ext_vector_type(8)))  _Float16 v8h;
typedef __attribute__((ext_vector_type(8)))  float    v8f;

#define N_TOK 8192
#define DMODEL 1024
#define KBLK 256          // flash key-block

__device__ __forceinline__ v8f wmma16(v16h a, v16h b, v8f c) {
  return __builtin_amdgcn_wmma_f32_16x16x32_f16(
      false, a, false, b, (short)0, c, false, false);
}

// A-matrix 16x32 f16 fragment (ISA 7.12.2): lane L row M=L%16, h=L/16;
// elems 0..7 <-> K=h*8+j, 8..15 <-> K=16+h*8+j. p = row base + k0.
__device__ __forceinline__ v16h frag_a_f16(const _Float16* p, int h) {
  v16h r;
  ((int4*)&r)[0] = *(const int4*)(p + h * 8);
  ((int4*)&r)[1] = *(const int4*)(p + 16 + h * 8);
  return r;
}

// B-matrix 32x16 f16 fragment: lane L holds column N=L%16;
// elems 0..7 <-> K=h*8+j, 8..15 <-> K=16+h*8+j.
__device__ __forceinline__ v16h frag_b_f16(const _Float16* p, int h) {
  v16h r;
  ((int4*)&r)[0] = *(const int4*)(p + h * 8);
  ((int4*)&r)[1] = *(const int4*)(p + 16 + h * 8);
  return r;
}

// ---------------------------------------------------------------------------
// Kernel 0: bulk fp32 -> f16 conversion (bandwidth-trivial).
// ---------------------------------------------------------------------------
__global__ __launch_bounds__(256) void cvt_f16_kernel(
    const float* __restrict__ src, _Float16* __restrict__ dst, int n) {
  const int i = (blockIdx.x * 256 + threadIdx.x) * 8;
  if (i < n) {
    const float4 a = *(const float4*)(src + i);
    const float4 b = *(const float4*)(src + i + 4);
    v8h o;
    o[0] = (_Float16)a.x; o[1] = (_Float16)a.y;
    o[2] = (_Float16)a.z; o[3] = (_Float16)a.w;
    o[4] = (_Float16)b.x; o[5] = (_Float16)b.y;
    o[6] = (_Float16)b.z; o[7] = (_Float16)b.w;
    *(v8h*)(dst + i) = o;
  }
}

// ---------------------------------------------------------------------------
// Kernel 1: Y = A @ W^T + b (f16 operands), f16 out (transposed for V).
// Block tile 128(M) x 128(N), waves 4(M) x 2(N); wave tile 32x64:
// 2 A-frags x 4 B-frags -> 8 accums, 1.5 b128-loads per WMMA.
// Fully unrolled, ping-pong buffer sets, immediate-offset addressing.
// ---------------------------------------------------------------------------
__global__ __launch_bounds__(256) void qkv_gemm_kernel(
    const _Float16* __restrict__ A, const _Float16* __restrict__ W,
    const float* __restrict__ bias, _Float16* __restrict__ dst, int vtrans) {
  const int lane = threadIdx.x & 31;
  const int wave = threadIdx.x >> 5;
  const int l16  = lane & 15;
  const int h    = lane >> 4;
  const int mw   = wave & 3;
  const int nw   = wave >> 2;
  const int row0 = blockIdx.x * 128 + mw * 32;
  const int col0 = blockIdx.y * 128 + nw * 64;

  const _Float16* arow[2];
  arow[0] = A + (size_t)(row0 + l16) * DMODEL;
  arow[1] = A + (size_t)(row0 + 16 + l16) * DMODEL;
  const _Float16* brow[4];
#pragma unroll
  for (int c = 0; c < 4; ++c)
    brow[c] = W + (size_t)(col0 + c * 16 + l16) * DMODEL;

  v8f acc[2][4] = {};
  v16h a0[2], b0[4], a1[2], b1[4];
#pragma unroll
  for (int m = 0; m < 2; ++m) a0[m] = frag_a_f16(arow[m], h);
#pragma unroll
  for (int c = 0; c < 4; ++c) b0[c] = frag_b_f16(brow[c], h);

#pragma unroll
  for (int k0 = 0; k0 < DMODEL; k0 += 64) {
#pragma unroll
    for (int m = 0; m < 2; ++m) a1[m] = frag_a_f16(arow[m] + k0 + 32, h);
#pragma unroll
    for (int c = 0; c < 4; ++c) b1[c] = frag_b_f16(brow[c] + k0 + 32, h);
#pragma unroll
    for (int m = 0; m < 2; ++m)
#pragma unroll
      for (int c = 0; c < 4; ++c) acc[m][c] = wmma16(a0[m], b0[c], acc[m][c]);
    if (k0 + 64 < DMODEL) {   // folds away on the last unrolled iteration
#pragma unroll
      for (int m = 0; m < 2; ++m) a0[m] = frag_a_f16(arow[m] + k0 + 64, h);
#pragma unroll
      for (int c = 0; c < 4; ++c) b0[c] = frag_b_f16(brow[c] + k0 + 64, h);
    }
#pragma unroll
    for (int m = 0; m < 2; ++m)
#pragma unroll
      for (int c = 0; c < 4; ++c) acc[m][c] = wmma16(a1[m], b1[c], acc[m][c]);
  }

  // Epilogue: C layout = VGPR r: rows r (lanes 0-15) / r+8 (lanes 16-31).
#pragma unroll
  for (int m = 0; m < 2; ++m)
#pragma unroll
    for (int c = 0; c < 4; ++c) {
      const int col = col0 + c * 16 + l16;
      const float bv = bias[col];
#pragma unroll
      for (int r = 0; r < 8; ++r) {
        const int row = row0 + m * 16 + r + h * 8;
        const float y = acc[m][c][r] + bv;
        if (vtrans) dst[(size_t)col * N_TOK + row] = (_Float16)y;   // V^T
        else        dst[(size_t)row * DMODEL + col] = (_Float16)y;
      }
    }
}

// ---------------------------------------------------------------------------
// Kernel 2: fused flash attention, key-block 256.
// Block = 32 Q rows, 8 waves: mw = wave&1 (16 rows), nw = wave>>1 (0..3):
// phase 1 each wave owns 64 score cols (4 accums); phase 2 a 256-wide D
// slice (16 accums). Q block lives in LDS; P bounced through LDS as f16.
// ---------------------------------------------------------------------------
__global__ __launch_bounds__(256) void flash_attn_kernel(
    const _Float16* __restrict__ Qf, const _Float16* __restrict__ Kf,
    const _Float16* __restrict__ Vt, float* __restrict__ Out) {
  __shared__ _Float16 Qlds[32][DMODEL + 8];  // padded row stride
  __shared__ _Float16 Plds[32][KBLK + 8];
  __shared__ float smax[2][4][16];
  __shared__ float ssum[2][4][16];

  const int tid  = threadIdx.x;
  const int lane = tid & 31;
  const int wave = tid >> 5;
  const int l16  = lane & 15;
  const int h    = lane >> 4;
  const int mw   = wave & 1;
  const int nw   = wave >> 1;
  const int row0 = blockIdx.x * 32 + mw * 16;
  const int blk0 = blockIdx.x * 32;

  // Stage 32x1024 Q block into LDS once (reused for all 32 key blocks).
  for (int ci = tid; ci < 32 * (DMODEL / 8); ci += 256) {
    const int r = ci >> 7, cc = ci & 127;
    *(int4*)&Qlds[r][cc * 8] =
        *(const int4*)(Qf + (size_t)(blk0 + r) * DMODEL + cc * 8);
  }
  __syncthreads();

  float m_run[8], l_run[8];
#pragma unroll
  for (int r = 0; r < 8; ++r) { m_run[r] = -INFINITY; l_run[r] = 0.f; }
  const v8f vzero = {};
  v8f o[16];
#pragma unroll
  for (int c = 0; c < 16; ++c) o[c] = vzero;

  for (int j = 0; j < N_TOK / KBLK; ++j) {
    const int key0 = j * KBLK;

    // Warm caches for the next key block (global_prefetch_b8).
    const int keyn = (key0 + KBLK) & (N_TOK - 1);
    __builtin_prefetch(Kf + (size_t)(keyn + tid) * DMODEL, 0, 1);
    __builtin_prefetch(Vt + (size_t)(tid * 4) * N_TOK + keyn, 0, 1);

    // ---- Phase 1: S = Q @ K^T for this wave's 64 key columns ----
    const _Float16* qrow = &Qlds[mw * 16 + l16][0];
    const _Float16* krow[4];
#pragma unroll
    for (int c = 0; c < 4; ++c)
      krow[c] = Kf + (size_t)(key0 + nw * 64 + c * 16 + l16) * DMODEL;

    v8f s[4];
#pragma unroll
    for (int c = 0; c < 4; ++c) s[c] = vzero;
    v16h a0 = frag_a_f16(qrow, h), a1;
    v16h b0[4], b1[4];
#pragma unroll
    for (int c = 0; c < 4; ++c) b0[c] = frag_b_f16(krow[c], h);
#pragma unroll
    for (int k0 = 0; k0 < DMODEL; k0 += 64) {
      a1 = frag_a_f16(qrow + k0 + 32, h);
#pragma unroll
      for (int c = 0; c < 4; ++c) b1[c] = frag_b_f16(krow[c] + k0 + 32, h);
#pragma unroll
      for (int c = 0; c < 4; ++c) s[c] = wmma16(a0, b0[c], s[c]);
      if (k0 + 64 < DMODEL) {
        a0 = frag_a_f16(qrow + k0 + 64, h);
#pragma unroll
        for (int c = 0; c < 4; ++c) b0[c] = frag_b_f16(krow[c] + k0 + 64, h);
      }
#pragma unroll
      for (int c = 0; c < 4; ++c) s[c] = wmma16(a1, b1[c], s[c]);
    }

    // Row max within wave (rows live in 16-lane halves of each acc VGPR).
    float rm[8];
#pragma unroll
    for (int r = 0; r < 8; ++r) {
      float t = fmaxf(fmaxf(s[0][r], s[1][r]), fmaxf(s[2][r], s[3][r]));
      t = fmaxf(t, __shfl_xor(t, 1, 16));
      t = fmaxf(t, __shfl_xor(t, 2, 16));
      t = fmaxf(t, __shfl_xor(t, 4, 16));
      t = fmaxf(t, __shfl_xor(t, 8, 16));
      rm[r] = t;
    }
    if (l16 == 0) {
#pragma unroll
      for (int r = 0; r < 8; ++r) smax[mw][nw][h * 8 + r] = rm[r];
    }
    __syncthreads();

    float scale[8];
#pragma unroll
    for (int r = 0; r < 8; ++r) {
      float t = fmaxf(fmaxf(smax[mw][0][h * 8 + r], smax[mw][1][h * 8 + r]),
                      fmaxf(smax[mw][2][h * 8 + r], smax[mw][3][h * 8 + r]));
      const float mnew = fmaxf(m_run[r], t);
      scale[r] = __expf(m_run[r] - mnew);
      m_run[r] = mnew;
    }

    // P = exp(S - m), partial row sums, stash P tile in LDS as f16.
    float rs[8];
#pragma unroll
    for (int r = 0; r < 8; ++r) {
      float t = 0.f;
#pragma unroll
      for (int c = 0; c < 4; ++c) {
        const float p = __expf(s[c][r] - m_run[r]);
        s[c][r] = p;
        t += p;
        Plds[mw * 16 + r + h * 8][nw * 64 + c * 16 + l16] = (_Float16)p;
      }
      t += __shfl_xor(t, 1, 16);
      t += __shfl_xor(t, 2, 16);
      t += __shfl_xor(t, 4, 16);
      t += __shfl_xor(t, 8, 16);
      rs[r] = t;
    }
    if (l16 == 0) {
#pragma unroll
      for (int r = 0; r < 8; ++r) ssum[mw][nw][h * 8 + r] = rs[r];
    }
    __syncthreads();

#pragma unroll
    for (int r = 0; r < 8; ++r) {
      const float tot = ssum[mw][0][h * 8 + r] + ssum[mw][1][h * 8 + r] +
                        ssum[mw][2][h * 8 + r] + ssum[mw][3][h * 8 + r];
      l_run[r] = l_run[r] * scale[r] + tot;
    }
#pragma unroll
    for (int c = 0; c < 16; ++c)
#pragma unroll
      for (int r = 0; r < 8; ++r) o[c][r] *= scale[r];

    // ---- Phase 2: O += P @ V (V^T -> contiguous B-frags) ----
    v16h pa = frag_a_f16(&Plds[mw * 16 + l16][0], h);
#pragma unroll
    for (int kk = 0; kk < KBLK / 32; ++kk) {
      v16h pa_n = pa;
      if (kk + 1 < KBLK / 32)
        pa_n = frag_a_f16(&Plds[mw * 16 + l16][(kk + 1) * 32], h);
      const _Float16* vbase = Vt + key0 + kk * 32;
      v16h bb[2];
      bb[0] = frag_b_f16(vbase + (size_t)(nw * 256 + l16) * N_TOK, h);
#pragma unroll
      for (int c = 0; c < 16; ++c) {
        if (c + 1 < 16)
          bb[(c + 1) & 1] = frag_b_f16(
              vbase + (size_t)(nw * 256 + (c + 1) * 16 + l16) * N_TOK, h);
        o[c] = wmma16(pa, bb[c & 1], o[c]);
      }
      pa = pa_n;
    }
    __syncthreads();
  }

  float invl[8];
#pragma unroll
  for (int r = 0; r < 8; ++r) invl[r] = 1.0f / l_run[r];
#pragma unroll
  for (int c = 0; c < 16; ++c) {
    const int col = nw * 256 + c * 16 + l16;
#pragma unroll
    for (int r = 0; r < 8; ++r) {
      const int row = row0 + r + h * 8;
      Out[(size_t)row * DMODEL + col] = o[c][r] * invl[r];
    }
  }
}

// ---------------------------------------------------------------------------
extern "C" void kernel_launch(void* const* d_in, const int* in_sizes, int n_in,
                              void* d_out, int out_size, void* d_ws,
                              size_t ws_size, hipStream_t stream) {
  (void)in_sizes; (void)n_in; (void)out_size; (void)ws_size;
  const float* E  = (const float*)d_in[0];
  const float* Wq = (const float*)d_in[1];
  const float* bq = (const float*)d_in[2];
  const float* Wk = (const float*)d_in[3];
  const float* bk = (const float*)d_in[4];
  const float* Wv = (const float*)d_in[5];
  const float* bv = (const float*)d_in[6];
  float* Out = (float*)d_out;

  // Workspace (f16): E 16MB | Wq,Wk,Wv 2MB each | Q 16MB | K 16MB | V^T 16MB
  _Float16* Ef  = (_Float16*)d_ws;
  _Float16* Wqf = Ef  + (size_t)N_TOK * DMODEL;
  _Float16* Wkf = Wqf + (size_t)DMODEL * DMODEL;
  _Float16* Wvf = Wkf + (size_t)DMODEL * DMODEL;
  _Float16* Qf  = Wvf + (size_t)DMODEL * DMODEL;
  _Float16* Kf  = Qf  + (size_t)N_TOK * DMODEL;
  _Float16* Vt  = Kf  + (size_t)N_TOK * DMODEL;

  const int nE = N_TOK * DMODEL, nW = DMODEL * DMODEL;
  cvt_f16_kernel<<<nE / 2048, 256, 0, stream>>>(E,  Ef,  nE);
  cvt_f16_kernel<<<nW / 2048, 256, 0, stream>>>(Wq, Wqf, nW);
  cvt_f16_kernel<<<nW / 2048, 256, 0, stream>>>(Wk, Wkf, nW);
  cvt_f16_kernel<<<nW / 2048, 256, 0, stream>>>(Wv, Wvf, nW);

  dim3 g1(N_TOK / 128, DMODEL / 128);
  qkv_gemm_kernel<<<g1, 256, 0, stream>>>(Ef, Wqf, bq, Qf, 0);
  qkv_gemm_kernel<<<g1, 256, 0, stream>>>(Ef, Wkf, bk, Kf, 0);
  qkv_gemm_kernel<<<g1, 256, 0, stream>>>(Ef, Wvf, bv, Vt, 1);
  flash_attn_kernel<<<N_TOK / 32, 256, 0, stream>>>(Qf, Kf, Vt, Out);
}
